// SpatialConv_188978561193
// MI455X (gfx1250) — compile-verified, matrix operand order checked by reading, not run on hardware
//
#include <hip/hip_runtime.h>
#include <math.h>

typedef __attribute__((ext_vector_type(16))) __bf16 v16bf;
typedef __attribute__((ext_vector_type(8)))  __bf16 v8bf;
typedef __attribute__((ext_vector_type(8)))  float  v8f;

// ---------------- sizes ----------------
#define NN   10000      // nodes
#define EE   80000      // edges
#define E2E_ 160000
#define N2N_ 160000
#define HC   512        // H*C
#define CK   128        // feature dim / K for GAT gemms
#define FCK  288        // padded K for fused edge MLP (128+128+16+pad)
#define NT   4          // N-tiles (16 cols each) per wave

// load a 16-element bf16 WMMA fragment: two aligned 16B runs at kb and kb+16
__device__ __forceinline__ v16bf load_frag_bf16(const __bf16* __restrict__ p, int kb) {
  v8bf lo = *(const v8bf*)(p + kb);
  v8bf hi = *(const v8bf*)(p + kb + 16);
  v16bf f;
  #pragma unroll
  for (int j = 0; j < 8; ++j) { f[j] = lo[j]; f[j + 8] = hi[j]; }
  return f;
}

// ================= prep: weights -> transposed bf16 [N][K] =================
__global__ void prep_wt(const float* __restrict__ W, __bf16* __restrict__ Wt,
                        int K, int Nout) {
  int i = blockIdx.x * blockDim.x + threadIdx.x;
  if (i >= K * Nout) return;
  int n = i / K, k = i - n * K;
  Wt[i] = (__bf16)W[(size_t)k * Nout + n];
}

// fused fc weight: wt_fc[n][k]: k<256 -> fc_W[k][n]; 256..271 -> (W_emb @ fc_W[256:288])[k-256][n]; else 0
__global__ void prep_fc(const float* __restrict__ fcW, const float* __restrict__ Wemb,
                        __bf16* __restrict__ wt_fc) {
  int i = blockIdx.x * blockDim.x + threadIdx.x;
  if (i >= 128 * FCK) return;
  int n = i / FCK, k = i - n * FCK;
  float v;
  if (k < 256) {
    v = fcW[(size_t)k * 128 + n];
  } else if (k < 272) {
    int j = k - 256;
    v = 0.f;
    #pragma unroll
    for (int q = 0; q < 32; ++q)
      v += Wemb[j * 32 + q] * fcW[(size_t)(256 + q) * 128 + n];
  } else {
    v = 0.f;
  }
  wt_fc[i] = (__bf16)v;
}

// ============ dense GEMM: Y[nrows,Nout] = Xb[nrows,K](bf16) @ Wt^T, Wt [Nout][K] bf16 ============
// one wave -> 16 x (16*NT) output tile; nrows % 16 == 0, Nout % (16*NT) == 0
__global__ void gemm_bf16_wmma(const __bf16* __restrict__ Xb, const __bf16* __restrict__ Wt,
                               float* __restrict__ Y, int nrows, int K, int Nout) {
  int wave = (int)((blockIdx.x * (size_t)blockDim.x + threadIdx.x) >> 5);
  int lane = threadIdx.x & 31;
  int ngrp = Nout / (16 * NT);
  int ntM  = nrows >> 4;
  if (wave >= ntM * ngrp) return;
  int mt = wave / ngrp, ng = wave - mt * ngrp;
  int row = mt * 16 + (lane & 15);
  int colbase = ng * (16 * NT) + (lane & 15);
  int krowoff = (lane >> 4) << 3;            // 0 or 8
  const __bf16* xp = Xb + (size_t)row * K;
  const __bf16* wp = Wt + (size_t)colbase * K;
  v8f zero = {0.f, 0.f, 0.f, 0.f, 0.f, 0.f, 0.f, 0.f};
  v8f acc[NT];
  #pragma unroll
  for (int t = 0; t < NT; ++t) acc[t] = zero;
  for (int k0 = 0; k0 < K; k0 += 32) {
    int kb = k0 + krowoff;
    v16bf a = load_frag_bf16(xp, kb);
    #pragma unroll
    for (int t = 0; t < NT; ++t) {
      v16bf b = load_frag_bf16(wp + (size_t)t * 16 * K, kb);
      acc[t] = __builtin_amdgcn_wmma_f32_16x16x32_bf16(false, a, false, b, (short)0,
                                                       acc[t], false, false);
    }
  }
  int m0 = mt * 16 + ((lane >> 4) << 3);
  #pragma unroll
  for (int t = 0; t < NT; ++t) {
    int col = colbase + t * 16;
    #pragma unroll
    for (int v = 0; v < 8; ++v)
      Y[(size_t)(m0 + v) * Nout + col] = acc[t][v];
  }
}

// ============ fused edge MLP: bf16 out; gathered virtual row [nf[s]|nf[d]|attr|0], K=288 ============
__global__ void edge_mlp_wmma(const float* __restrict__ nodef, const float* __restrict__ eattr,
                              const int* __restrict__ esrc, const int* __restrict__ edst,
                              const __bf16* __restrict__ wt_fc, const float* __restrict__ fcb,
                              __bf16* __restrict__ Y, int E) {
  const int K = FCK, Nout = 128;
  int wave = (int)((blockIdx.x * (size_t)blockDim.x + threadIdx.x) >> 5);
  int lane = threadIdx.x & 31;
  int ngrp = Nout / (16 * NT);               // 2
  int ntM  = E >> 4;
  if (wave >= ntM * ngrp) return;
  int mt = wave / ngrp, ng = wave - mt * ngrp;
  int m = mt * 16 + (lane & 15);
  int s = esrc[m], d = edst[m];
  const float* ps = nodef + (size_t)s * 128;
  const float* pd = nodef + (size_t)d * 128;
  const float* pe = eattr + (size_t)m * 16;
  int colbase = ng * (16 * NT) + (lane & 15);
  int krowoff = (lane >> 4) << 3;
  const __bf16* wp = wt_fc + (size_t)colbase * K;
  v8f zero = {0.f, 0.f, 0.f, 0.f, 0.f, 0.f, 0.f, 0.f};
  v8f acc[NT];
  #pragma unroll
  for (int t = 0; t < NT; ++t) acc[t] = zero;
  for (int k0 = 0; k0 < K; k0 += 32) {
    v16bf a;
    #pragma unroll
    for (int half = 0; half < 2; ++half) {
      int kb = k0 + krowoff + half * 16;     // 8-aligned; never crosses a region boundary
      #pragma unroll
      for (int j = 0; j < 8; ++j) {
        int k = kb + j;
        float v;
        if (k < 128)      v = ps[k];
        else if (k < 256) v = pd[k - 128];
        else if (k < 272) v = pe[k - 256];
        else              v = 0.f;
        a[half * 8 + j] = (__bf16)v;
      }
    }
    int kb = k0 + krowoff;
    #pragma unroll
    for (int t = 0; t < NT; ++t) {
      v16bf b = load_frag_bf16(wp + (size_t)t * 16 * K, kb);
      acc[t] = __builtin_amdgcn_wmma_f32_16x16x32_bf16(false, a, false, b, (short)0,
                                                       acc[t], false, false);
    }
  }
  int m0 = mt * 16 + ((lane >> 4) << 3);
  #pragma unroll
  for (int t = 0; t < NT; ++t) {
    int col = colbase + t * 16;
    float bcol = fcb[col];
    #pragma unroll
    for (int v = 0; v < 8; ++v) {
      float r = fmaxf(acc[t][v] + bcol, 0.f);
      Y[(size_t)(m0 + v) * 128 + col] = (__bf16)r;
    }
  }
}

// ================= GATv2 pieces (fp32) =================
__device__ __forceinline__ unsigned enc_f(float f) {
  unsigned u = __float_as_uint(f);
  return ((int)u >= 0) ? (u | 0x80000000u) : ~u;   // monotone float->uint
}
__device__ __forceinline__ float dec_f(unsigned u) {
  return __uint_as_float((u & 0x80000000u) ? (u & 0x7FFFFFFFu) : ~u);
}

// one wave per message; logits[m][h] = sum_c leaky_relu(xl[s]+xr[d], 0.2) * att[h][c]
__global__ void gat_logits(const float* __restrict__ xl, const float* __restrict__ xr,
                           const int* __restrict__ src, const int* __restrict__ dst,
                           const float* __restrict__ att, float* __restrict__ logits,
                           int Eg, int M) {
  int wid  = (int)((blockIdx.x * (size_t)blockDim.x + threadIdx.x) >> 5);
  int lane = threadIdx.x & 31;
  if (wid >= M) return;
  int s, d;
  if (wid < Eg) { s = src[wid]; d = dst[wid]; } else { s = d = wid - Eg; }
  const float* pl = xl + (size_t)s * HC;
  const float* pr = xr + (size_t)d * HC;
  #pragma unroll
  for (int h = 0; h < 4; ++h) {
    float acc = 0.f;
    const float* pa = att + h * 128;
    #pragma unroll
    for (int t = 0; t < 4; ++t) {
      int c = lane + t * 32;
      float v = pl[h * 128 + c] + pr[h * 128 + c];
      v = (v > 0.f) ? v : 0.2f * v;
      acc += v * pa[c];
    }
    #pragma unroll
    for (int off = 16; off > 0; off >>= 1) acc += __shfl_xor(acc, off, 32);
    if (lane == 0) logits[(size_t)wid * 4 + h] = acc;
  }
}

__global__ void seg_max(const float* __restrict__ logits, const int* __restrict__ dst,
                        unsigned* __restrict__ smax, int Eg, int M) {
  int i = blockIdx.x * blockDim.x + threadIdx.x;
  if (i >= M * 4) return;
  int m = i >> 2, h = i & 3;
  int d = (m < Eg) ? dst[m] : m - Eg;
  atomicMax(&smax[d * 4 + h], enc_f(logits[i]));
}

__global__ void seg_expsum(float* __restrict__ logits, const int* __restrict__ dst,
                           const unsigned* __restrict__ smax, float* __restrict__ ssum,
                           int Eg, int M) {
  int i = blockIdx.x * blockDim.x + threadIdx.x;
  if (i >= M * 4) return;
  int m = i >> 2, h = i & 3;
  int d = (m < Eg) ? dst[m] : m - Eg;
  float e = expf(logits[i] - dec_f(smax[d * 4 + h]));
  logits[i] = e;                       // reuse buffer as un-normalized alpha
  atomicAdd(&ssum[d * 4 + h], e);
}

// one 128-thread block per message: agg[d][h][c] += xl[s][h][c] * alpha
__global__ void seg_agg(const float* __restrict__ xl, const float* __restrict__ alpha,
                        const float* __restrict__ ssum, const int* __restrict__ src,
                        const int* __restrict__ dst, float* __restrict__ agg,
                        int Eg, int M) {
  int m = blockIdx.x;
  if (m >= M) return;
  int c = threadIdx.x;
  int s, d;
  if (m < Eg) { s = src[m]; d = dst[m]; } else { s = d = m - Eg; }
  #pragma unroll
  for (int h = 0; h < 4; ++h) {
    float a = alpha[(size_t)m * 4 + h] / (ssum[d * 4 + h] + 1e-16f);
    atomicAdd(&agg[(size_t)d * HC + h * 128 + c],
              xl[(size_t)s * HC + h * 128 + c] * a);
  }
}

// out[i][c] = mean_h(agg[i][h][c] + b[h][c])
__global__ void finalize_mean(const float* __restrict__ agg, const float* __restrict__ bias,
                              float* __restrict__ out, int n) {
  int i = blockIdx.x * blockDim.x + threadIdx.x;
  if (i >= n * 128) return;
  int r = i >> 7, c = i & 127;
  const float* p = agg + (size_t)r * HC;
  float v = p[c] + p[128 + c] + p[256 + c] + p[384 + c]
          + bias[c] + bias[128 + c] + bias[256 + c] + bias[384 + c];
  out[i] = 0.25f * v;
}

__global__ void zero_f32(float* __restrict__ p, size_t n) {
  size_t i = blockIdx.x * (size_t)blockDim.x + threadIdx.x;
  if (i < n) p[i] = 0.f;
}

__global__ void cvt_f32_bf16(const float* __restrict__ src, __bf16* __restrict__ dst, size_t n) {
  size_t i = blockIdx.x * (size_t)blockDim.x + threadIdx.x;
  if (i < n) dst[i] = (__bf16)src[i];
}

// ================= host =================
extern "C" void kernel_launch(void* const* d_in, const int* in_sizes, int n_in,
                              void* d_out, int out_size, void* d_ws, size_t ws_size,
                              hipStream_t stream) {
  const float* node_feat = (const float*)d_in[0];
  const float* edge_attr = (const float*)d_in[1];
  const float* W_emb     = (const float*)d_in[2];
  const float* fc_W      = (const float*)d_in[3];
  const float* fc_b      = (const float*)d_in[4];
  const float* Wl1  = (const float*)d_in[5];
  const float* Wr1  = (const float*)d_in[6];
  const float* att1 = (const float*)d_in[7];
  const float* b1   = (const float*)d_in[8];
  const float* Wl2  = (const float*)d_in[9];
  const float* Wr2  = (const float*)d_in[10];
  const float* att2 = (const float*)d_in[11];
  const float* b2   = (const float*)d_in[12];
  const int* eidx = (const int*)d_in[13];    // [2, EE]
  const int* e2e  = (const int*)d_in[14];    // [2, E2E]
  const int* n2n  = (const int*)d_in[15];    // [2, N2N]

  char* ws = (char*)d_ws;
  size_t off = 0;
  auto carve = [&](size_t bytes) -> void* {
    void* p = (void*)(ws + off);
    off += (bytes + 255) & ~(size_t)255;
    return p;
  };
  __bf16* wt_fc = (__bf16*)carve((size_t)128 * FCK * 2);
  __bf16* wt_l1 = (__bf16*)carve((size_t)HC * CK * 2);
  __bf16* wt_r1 = (__bf16*)carve((size_t)HC * CK * 2);
  __bf16* wt_l2 = (__bf16*)carve((size_t)HC * CK * 2);
  __bf16* wt_r2 = (__bf16*)carve((size_t)HC * CK * 2);
  __bf16* X1  = (__bf16*)carve((size_t)EE * 128 * 2);            // edge_feat (bf16, GEMM input)
  __bf16* X2  = (__bf16*)carve((size_t)(NN + EE) * 128 * 2);     // node_edge_feat (bf16)
  float* xl   = (float*)carve((size_t)(NN + EE) * HC * 4);
  float* xr   = (float*)carve((size_t)(NN + EE) * HC * 4);
  float* agg  = (float*)carve((size_t)(NN + EE) * HC * 4);
  float*    alpha = (float*)carve((size_t)(N2N_ + NN + EE) * 4 * 4);
  unsigned* smax  = (unsigned*)carve((size_t)(NN + EE) * 4 * 4);
  float*    ssum  = (float*)carve((size_t)(NN + EE) * 4 * 4);

  float* out_node = (float*)d_out;                               // [NN+EE, 128]
  float* out_edge = out_node + (size_t)(NN + EE) * 128;          // [EE, 128]

  // ---- prep weights (bf16, transposed) ----
  prep_wt<<<(HC * CK + 255) / 256, 256, 0, stream>>>(Wl1, wt_l1, CK, HC);
  prep_wt<<<(HC * CK + 255) / 256, 256, 0, stream>>>(Wr1, wt_r1, CK, HC);
  prep_wt<<<(HC * CK + 255) / 256, 256, 0, stream>>>(Wl2, wt_l2, CK, HC);
  prep_wt<<<(HC * CK + 255) / 256, 256, 0, stream>>>(Wr2, wt_r2, CK, HC);
  prep_fc<<<(128 * FCK + 255) / 256, 256, 0, stream>>>(fc_W, W_emb, wt_fc);

  // ---- stage 1: fused edge MLP (WMMA, bf16 output) ----
  {
    int waves = (EE / 16) * (128 / (16 * NT));
    edge_mlp_wmma<<<(waves * 32 + 255) / 256, 256, 0, stream>>>(
        node_feat, edge_attr, eidx, eidx + EE, wt_fc, fc_b, X1, EE);
  }

  auto run_gat = [&](const __bf16* Xb, int nrows, const __bf16* wl, const __bf16* wr,
                     const float* att, const float* bias, const int* src,
                     const int* dst, int Eg, float* out) {
    int waves = (nrows / 16) * (HC / (16 * NT));
    gemm_bf16_wmma<<<(waves * 32 + 255) / 256, 256, 0, stream>>>(Xb, wl, xl, nrows, CK, HC);
    gemm_bf16_wmma<<<(waves * 32 + 255) / 256, 256, 0, stream>>>(Xb, wr, xr, nrows, CK, HC);
    size_t nz = (size_t)nrows * HC;
    zero_f32<<<(unsigned)((nz + 255) / 256), 256, 0, stream>>>(agg, nz);
    zero_f32<<<(nrows * 4 + 255) / 256, 256, 0, stream>>>((float*)smax, (size_t)nrows * 4);
    zero_f32<<<(nrows * 4 + 255) / 256, 256, 0, stream>>>(ssum, (size_t)nrows * 4);
    int M = Eg + nrows;
    gat_logits<<<((size_t)M * 32 + 255) / 256, 256, 0, stream>>>(xl, xr, src, dst, att, alpha, Eg, M);
    seg_max<<<(M * 4 + 255) / 256, 256, 0, stream>>>(alpha, dst, smax, Eg, M);
    seg_expsum<<<(M * 4 + 255) / 256, 256, 0, stream>>>(alpha, dst, smax, ssum, Eg, M);
    seg_agg<<<M, 128, 0, stream>>>(xl, alpha, ssum, src, dst, agg, Eg, M);
    finalize_mean<<<(nrows * 128 + 255) / 256, 256, 0, stream>>>(agg, bias, out, nrows);
  };

  // ---- stage 2: GATv2 over edge graph -> updated_edge_feat (fp32 in d_out) ----
  run_gat(X1, EE, wt_l1, wt_r1, att1, b1, e2e, e2e + E2E_, E2E_, out_edge);

  // ---- stage 3: X2 = concat(node_feat, updated_edge_feat) as bf16; GATv2 -> updated_node_feat ----
  {
    size_t n0 = (size_t)NN * 128, n1 = (size_t)EE * 128;
    cvt_f32_bf16<<<(unsigned)((n0 + 255) / 256), 256, 0, stream>>>(node_feat, X2, n0);
    cvt_f32_bf16<<<(unsigned)((n1 + 255) / 256), 256, 0, stream>>>(out_edge, X2 + n0, n1);
  }
  run_gat(X2, NN + EE, wt_l2, wt_r2, att2, b2, n2n, n2n + N2N_, N2N_, out_node);
}
